// VectorQuantizer_40742059770460
// MI455X (gfx1250) — matrix operand order repaired and checked by reading, not compiled, not run on hardware
//
#include <hip/hip_runtime.h>

// CDNA5 / gfx1250 implementation of the multi-scale residual VQ reference.
// B=512, N=16, C=256, K=8192. All heavy math goes through v_wmma_f32_16x16x32_f16.

#define B_DIM 512
#define N_DIM 16
#define C_DIM 256
#define K_CODES 8192
#define BNC (B_DIM * N_DIM * C_DIM)          // 2,097,152
#define CODES_PER_WAVE 256                    // 16 tiles of 16 codes per wave
#define CHUNKS (K_CODES / CODES_PER_WAVE)     // 32 code chunks

typedef __attribute__((ext_vector_type(16))) _Float16 v16h;
typedef __attribute__((ext_vector_type(8)))  float    v8f;

// ---------------------------------------------------------------- init
__global__ void vq_init(const float* __restrict__ f,
                        float* __restrict__ fHat,     // d_out region
                        float* __restrict__ fRest,
                        float* __restrict__ accum) {
  int tid = blockIdx.x * blockDim.x + threadIdx.x;
  if (tid < BNC) {
    fHat[tid]  = 0.0f;
    fRest[tid] = f[tid];
  }
  if (tid == 0) *accum = 0.0f;
}

// ------------------------------------------------- emb -> f16 + e_sq[k]
__global__ void vq_prep_emb(const float* __restrict__ emb,
                            _Float16* __restrict__ embH,
                            float* __restrict__ eSq) {
  int k = blockIdx.x;            // K_CODES blocks
  int c = threadIdx.x;           // 256 threads
  float v = emb[(size_t)k * C_DIM + c];
  embH[(size_t)k * C_DIM + c] = (_Float16)v;
  float sq = v * v;
  #pragma unroll
  for (int m = 16; m; m >>= 1) sq += __shfl_xor(sq, m);
  __shared__ float smem[8];
  if ((c & 31) == 0) smem[c >> 5] = sq;
  __syncthreads();
  if (c == 0) {
    float t = 0.0f;
    #pragma unroll
    for (int i = 0; i < 8; ++i) t += smem[i];
    eSq[k] = t;
  }
}

// ------------------------------- area-pool f_rest -> f16 rows, reset keys
__global__ void vq_pool(const float* __restrict__ fRest,
                        _Float16* __restrict__ restH,
                        unsigned long long* __restrict__ minkey,
                        int pn) {
  int tid = blockIdx.x * blockDim.x + threadIdx.x;
  int total = B_DIM * pn * C_DIM;
  if (tid >= total) return;
  int c    = tid & (C_DIM - 1);
  int rowm = tid >> 8;                 // b*pn + p
  int b = rowm / pn;
  int p = rowm - b * pn;
  int s = (p * N_DIM) / pn;
  int e = ((p + 1) * N_DIM + pn - 1) / pn;   // ceil
  float sum = 0.0f;
  for (int n = s; n < e; ++n)
    sum += fRest[((size_t)b * N_DIM + n) * C_DIM + c];
  restH[tid] = (_Float16)(sum / (float)(e - s));
  if (c == 0) minkey[rowm] = 0xFFFFFFFFFFFFFFFFull;
}

// ------------------------------------ WMMA distance + argmin (the hot loop)
// One wave (32 threads) per (16-row block, 256-code chunk).
// A: rest rows (16x32 f16 per fragment), B: emb codes (32x16 f16), D: f32 16x16.
__global__ __launch_bounds__(32)
void vq_argmin_wmma(const _Float16* __restrict__ restH,  // [M, 256]
                    const _Float16* __restrict__ embH,   // [K, 256]
                    const float* __restrict__ eSq,       // [K]
                    unsigned long long* __restrict__ minkey) {
  const int wave  = blockIdx.x;
  const int rb    = wave / CHUNKS;       // 16-row block
  const int chunk = wave % CHUNKS;       // code chunk
  const int lane  = threadIdx.x;
  const int l15   = lane & 15;
  const int khalf = lane >> 4;

  // --- A fragments: row m = l15; k layout (ISA 7.12.2, 16-bit A 16x32):
  //   elements 0..7  -> k = kc*32 + khalf*8 + 0..7
  //   elements 8..15 -> k = kc*32 + 16 + khalf*8 + 0..7
  const int row = rb * 16 + l15;
  const _Float16* arow = restH + (size_t)row * C_DIM;
  v16h afrag[8];
  #pragma unroll
  for (int kc = 0; kc < 8; ++kc) {
    const _Float16* p0 = arow + kc * 32 + khalf * 8;
    union { uint4 q[2]; v16h v; } u;
    u.q[0] = *(const uint4*)(p0);
    u.q[1] = *(const uint4*)(p0 + 16);
    afrag[kc] = u.v;
  }

  float bestD[8];
  int   bestI[8];
  #pragma unroll
  for (int v = 0; v < 8; ++v) { bestD[v] = 3.4e38f; bestI[v] = 0; }

  const int code0 = chunk * CODES_PER_WAVE;
  for (int t = 0; t < CODES_PER_WAVE / 16; ++t) {
    const int code = code0 + t * 16 + l15;
    // --- B fragment: col n = l15; k = kc*32 + khalf*8 + 0..7, then kc*32 + 16 + khalf*8 + 0..7
    const _Float16* brow = embH + (size_t)code * C_DIM + khalf * 8;
    if (t + 1 < CODES_PER_WAVE / 16)
      __builtin_prefetch(brow + (size_t)16 * C_DIM, 0, 1);  // global_prefetch_b8
    v8f acc = {};
    #pragma unroll
    for (int kc = 0; kc < 8; ++kc) {
      union { uint4 q[2]; v16h v; } u;
      u.q[0] = *(const uint4*)(brow + kc * 32);
      u.q[1] = *(const uint4*)(brow + kc * 32 + 16);
      acc = __builtin_amdgcn_wmma_f32_16x16x32_f16(
          /*neg_a=*/false, afrag[kc], /*neg_b=*/false, u.v,
          /*c_mod=*/(short)0, acc, /*reuse_a=*/false, /*reuse_b=*/false);
    }
    // d = e_sq[code] - 2*(r . e)   (row norm constant -> dropped for argmin)
    const float es = eSq[code];
    #pragma unroll
    for (int v = 0; v < 8; ++v) {
      float d = es - 2.0f * acc[v];
      if (d < bestD[v]) { bestD[v] = d; bestI[v] = code; }  // codes ascend -> first-min
    }
  }

  // Butterfly min across the 16 lanes of each half (D layout: lane&15 = n,
  // VGPR v = row m with m = (lane>>4)*8 + v). xor masks 1..8 stay in-half.
  #pragma unroll
  for (int v = 0; v < 8; ++v) {
    float d = bestD[v];
    int   i = bestI[v];
    #pragma unroll
    for (int m = 8; m >= 1; m >>= 1) {
      float od = __shfl_xor(d, m);
      int   oi = __shfl_xor(i, m);
      if (od < d || (od == d && oi < i)) { d = od; i = oi; }
    }
    if (l15 == 0) {
      int mrow = rb * 16 + khalf * 8 + v;
      unsigned u = __float_as_uint(d);
      u ^= ((unsigned)(((int)u) >> 31)) | 0x80000000u;     // order-preserving map
      unsigned long long key = ((unsigned long long)u << 32) | (unsigned)i;
      atomicMin(minkey + mrow, key);                        // global_atomic_min_u64
    }
  }
}

// ---------------- gather + linear upsample + residual update + block sums
__global__ void vq_update(const float* __restrict__ f,
                          const float* __restrict__ emb,
                          const unsigned long long* __restrict__ minkey,
                          float* __restrict__ fHat,
                          float* __restrict__ fRest,
                          float* __restrict__ blockSums,
                          int pn) {
  int tid = blockIdx.x * blockDim.x + threadIdx.x;   // over BNC
  int c = tid & (C_DIM - 1);
  int n = (tid >> 8) & (N_DIM - 1);
  int b = tid >> 12;
  // F.interpolate(mode='linear', align_corners=False), pn -> 16
  float scale = (float)pn * (1.0f / 16.0f);
  float pos = fmaxf(((float)n + 0.5f) * scale - 0.5f, 0.0f);
  int i0 = (int)pos; if (i0 > pn - 1) i0 = pn - 1;
  int i1 = i0 + 1;   if (i1 > pn - 1) i1 = pn - 1;
  float frac = pos - (float)i0;
  int idx0 = (int)(unsigned)(minkey[b * pn + i0] & 0xFFFFFFFFull);
  int idx1 = (int)(unsigned)(minkey[b * pn + i1] & 0xFFFFFFFFull);
  float hv = (1.0f - frac) * emb[(size_t)idx0 * C_DIM + c]
           +         frac  * emb[(size_t)idx1 * C_DIM + c];
  float fh = fHat[tid] + hv;
  fHat[tid]  = fh;
  fRest[tid] -= hv;
  float diff = fh - f[tid];
  float sq = diff * diff;
  #pragma unroll
  for (int m = 16; m; m >>= 1) sq += __shfl_xor(sq, m);
  __shared__ float smem[8];
  if ((threadIdx.x & 31) == 0) smem[threadIdx.x >> 5] = sq;
  __syncthreads();
  if (threadIdx.x == 0) {
    float t = 0.0f;
    #pragma unroll
    for (int i = 0; i < 8; ++i) t += smem[i];
    blockSums[blockIdx.x] = t;   // no float atomics -> deterministic
  }
}

// ------------------------- deterministic reduce of block sums into accum
__global__ void vq_reduce(const float* __restrict__ blockSums,
                          float* __restrict__ accum, int nBlocks) {
  float s = 0.0f;
  for (int i = threadIdx.x; i < nBlocks; i += 256) s += blockSums[i];
  #pragma unroll
  for (int m = 16; m; m >>= 1) s += __shfl_xor(s, m);
  __shared__ float smem[8];
  if ((threadIdx.x & 31) == 0) smem[threadIdx.x >> 5] = s;
  __syncthreads();
  if (threadIdx.x == 0) {
    float t = 0.0f;
    #pragma unroll
    for (int i = 0; i < 8; ++i) t += smem[i];
    *accum += t;
  }
}

// --------------------------------------------------------- scalars out
__global__ void vq_finalize(const float* __restrict__ accum,
                            float* __restrict__ out) {
  float S = *accum;
  float qlat = S / ((float)BNC * (float)N_DIM);   // sum of per-stage means / N
  out[BNC + 0] = 0.25f * qlat;                    // commit
  out[BNC + 1] = qlat;                            // qlat
}

extern "C" void kernel_launch(void* const* d_in, const int* in_sizes, int n_in,
                              void* d_out, int out_size, void* d_ws, size_t ws_size,
                              hipStream_t stream) {
  const float* f   = (const float*)d_in[0];   // [B, N, C] fp32
  const float* emb = (const float*)d_in[1];   // [K, C]    fp32
  float* out = (float*)d_out;                 // [BNC floats][commit][qlat]

  // workspace carve-up (all offsets 256B-aligned)
  char* ws = (char*)d_ws;
  size_t off = 0;
  float* fRest = (float*)(ws + off);                off += (size_t)BNC * 4;          // 8 MB
  _Float16* restH = (_Float16*)(ws + off);          off += (size_t)K_CODES * C_DIM * 2; // 4 MB (max M=8192 rows)
  _Float16* embH  = (_Float16*)(ws + off);          off += (size_t)K_CODES * C_DIM * 2; // 4 MB
  float* eSq = (float*)(ws + off);                  off += (size_t)K_CODES * 4;
  unsigned long long* minkey = (unsigned long long*)(ws + off); off += (size_t)K_CODES * 8;
  float* blockSums = (float*)(ws + off);            off += (size_t)(BNC / 256) * 4;
  float* accum = (float*)(ws + off);                off += 256;

  vq_init<<<(BNC + 255) / 256, 256, 0, stream>>>(f, out, fRest, accum);
  vq_prep_emb<<<K_CODES, 256, 0, stream>>>(emb, embH, eSq);

  for (int pn = 1; pn <= N_DIM; ++pn) {
    const int M = B_DIM * pn;                 // always a multiple of 16
    const int total = M * C_DIM;
    vq_pool<<<(total + 255) / 256, 256, 0, stream>>>(fRest, restH, minkey, pn);
    const int waves = (M / 16) * CHUNKS;      // 1024..16384 waves
    vq_argmin_wmma<<<waves, 32, 0, stream>>>(restH, embH, eSq, minkey);
    vq_update<<<BNC / 256, 256, 0, stream>>>(f, emb, minkey, out, fRest, blockSums, pn);
    vq_reduce<<<1, 256, 0, stream>>>(blockSums, accum, BNC / 256);
  }
  vq_finalize<<<1, 1, 0, stream>>>(accum, out);
}
